// TransLayer_12910671692265
// MI455X (gfx1250) — matrix operand order repaired and checked by reading, not compiled, not run on hardware
//
#include <hip/hip_runtime.h>

#define BB 4
#define NN 8192
#define DD 512
#define HH 8
#define DH 64
#define BH 32          // BB*HH
#define MM 256         // landmarks
#define LL 32          // tokens per landmark
#define INNER 512
#define NQKV 1536
#define BN 32768       // BB*NN
#define KS 8           // key splits in flash kernel

typedef __attribute__((ext_vector_type(16))) _Float16 v16h;
typedef __attribute__((ext_vector_type(8)))  _Float16 v8h;
typedef __attribute__((ext_vector_type(8)))  float    v8f;

union F16x16 { v16h v; v8h p[2]; _Float16 h[16]; };
union F32x8  { v8f  v; float    f[8];  };
union H8     { v8h  v; _Float16 h[8];  };

__device__ inline v8f wmma_f16(v16h a, v16h b, v8f c) {
  return __builtin_amdgcn_wmma_f32_16x16x32_f16(false, a, false, b, (short)0, c, false, false);
}

__device__ inline void prefetch_ro(const void* p) { __builtin_prefetch(p, 0, 3); }

// A fragment (16x32 f16) at (row0,k0) from row-major A, leading dim ld.
// M = lane&15; half = lane>>4; 16 contiguous halves per lane (k = half*16 + s), same k order as B.
__device__ inline v16h load_a_frag(const _Float16* A, int ld, int row0, int k0) {
  int lane = threadIdx.x & 31;
  int m = lane & 15, half = lane >> 4;
  const _Float16* r = A + (size_t)(row0 + m) * ld + k0;
  F16x16 a;
  a.p[0] = *(const v8h*)(r + half * 16);
  a.p[1] = *(const v8h*)(r + half * 16 + 8);
  return a.v;
}

// B fragment where B = Mt^T, Mt row-major (N x K): 16 contiguous halfwords per lane
__device__ inline v16h load_bt_frag(const _Float16* Mt, int ld, int k0, int col0) {
  int lane = threadIdx.x & 31;
  int n = lane & 15, half = lane >> 4;
  return *(const v16h*)(Mt + (size_t)(col0 + n) * ld + k0 + half * 16);
}

// D store to f32 buffer (LDS): N = lane&15, M = half*8 + r
__device__ inline void store_d_f32(float* C, int ld, int row0, int col0, v8f acc) {
  int lane = threadIdx.x & 31;
  int n = lane & 15, half = lane >> 4;
  F32x8 a; a.v = acc;
#pragma unroll
  for (int r = 0; r < 8; ++r)
    C[(size_t)(row0 + half * 8 + r) * ld + col0 + n] = a.f[r];
}

// D store into TRANSPOSED f16 buffer Ct[N][M]: one packed b128 store per lane
__device__ inline void store_dT_f16(_Float16* Ct, int ldt, int row0, int col0, v8f acc, float scale) {
  int lane = threadIdx.x & 31;
  int n = lane & 15, half = lane >> 4;
  F32x8 a; a.v = acc;
  H8 o;
#pragma unroll
  for (int r = 0; r < 8; ++r) o.h[r] = (_Float16)(a.f[r] * scale);
  *(v8h*)(Ct + (size_t)(col0 + n) * ldt + row0 + half * 8) = o.v;
}

// Transposed store of (beta*I - P)
__device__ inline void store_dT_epi(_Float16* Ct, int ldt, int row0, int col0, v8f acc, float beta) {
  int lane = threadIdx.x & 31;
  int n = lane & 15, half = lane >> 4;
  int col = col0 + n;
  F32x8 a; a.v = acc;
  H8 o;
#pragma unroll
  for (int r = 0; r < 8; ++r) {
    int row = row0 + half * 8 + r;
    o.h[r] = (_Float16)(((row == col) ? beta : 0.0f) - a.f[r]);
  }
  *(v8h*)(Ct + (size_t)col * ldt + row0 + half * 8) = o.v;
}

// ---------------- LayerNorm (f32 in -> f16 normalized) ----------------
__global__ void k_layernorm(const float* __restrict__ x, const float* __restrict__ g,
                            const float* __restrict__ be, _Float16* __restrict__ xn) {
  __shared__ float red[256];
  int row = blockIdx.x;
  int t = threadIdx.x;
  const float* xr = x + (size_t)row * DD;
  float v0 = xr[t], v1 = xr[t + 256];
  red[t] = v0 + v1;
  __syncthreads();
  for (int s = 128; s > 0; s >>= 1) { if (t < s) red[t] += red[t + s]; __syncthreads(); }
  float mu = red[0] * (1.0f / DD);
  __syncthreads();
  float d0 = v0 - mu, d1 = v1 - mu;
  red[t] = d0 * d0 + d1 * d1;
  __syncthreads();
  for (int s = 128; s > 0; s >>= 1) { if (t < s) red[t] += red[t + s]; __syncthreads(); }
  float rinv = rsqrtf(red[0] * (1.0f / DD) + 1e-5f);
  xn[(size_t)row * DD + t]       = (_Float16)(d0 * rinv * g[t] + be[t]);
  xn[(size_t)row * DD + t + 256] = (_Float16)(d1 * rinv * g[t + 256] + be[t + 256]);
}

// ---------------- weight conversion f32 -> TRANSPOSED f16 ----------------
__global__ void k_cvt_w(const float* __restrict__ wq, const float* __restrict__ wo,
                        _Float16* __restrict__ wqT, _Float16* __restrict__ woT) {
  int i = blockIdx.x * 256 + threadIdx.x;
  if (i < DD * NQKV) {
    int r = i / NQKV, c = i % NQKV;
    wqT[(size_t)c * DD + r] = (_Float16)wq[i];
  }
  if (i < DD * DD) {
    int r = i / DD, c = i % DD;
    woT[(size_t)c * DD + r] = (_Float16)wo[i];
  }
}

// ---------------- QKV tile store helper ----------------
__device__ inline void qkv_store_tile(v8f acc, int row0, int col0,
                                      _Float16* __restrict__ q, _Float16* __restrict__ k,
                                      _Float16* __restrict__ vT) {
  int lane = threadIdx.x & 31;
  int n = lane & 15, half = lane >> 4;
  int col = col0 + n;
  int sec = col / INNER, c2 = col % INNER;
  int head = c2 / DH, dh = c2 % DH;
  F32x8 a; a.v = acc;
  if (sec == 2) {  // v: transposed layout, 8 consecutive tokens -> one b128 store
    H8 o;
#pragma unroll
    for (int r = 0; r < 8; ++r) o.h[r] = (_Float16)a.f[r];
    int gt0 = row0 + half * 8;
    int bi = gt0 >> 13, tok0 = gt0 & (NN - 1);
    *(v8h*)(vT + ((size_t)(bi * HH + head) * DH + dh) * NN + tok0) = o.v;
  } else {
    _Float16* dst = (sec == 0) ? q : k;
    float scale = (sec == 0) ? 0.125f : 1.0f;  // DIM_HEAD**-0.5 on q
#pragma unroll
    for (int r = 0; r < 8; ++r) {
      int gt = row0 + half * 8 + r;
      int bi = gt >> 13, tok = gt & (NN - 1);
      dst[(size_t)((bi * HH + head) * NN + tok) * DH + dh] = (_Float16)(a.f[r] * scale);
    }
  }
}

// ---------------- QKV GEMM (32x32 tile / wave): xn[32768,512] @ wq[512,1536] ----------------
__global__ void k_qkv_gemm(const _Float16* __restrict__ xn, const _Float16* __restrict__ wqT,
                           _Float16* __restrict__ q, _Float16* __restrict__ k,
                           _Float16* __restrict__ vT) {
  int wave = threadIdx.x >> 5;
  int tile = blockIdx.x * 8 + wave;
  const int TN = NQKV / 32;                 // 48
  int tm = tile / TN, tn = tile % TN;
  int lane = threadIdx.x & 31;
  int n = lane & 15;
  v8f a00 = {}, a01 = {}, a10 = {}, a11 = {};
#pragma unroll 4
  for (int k0 = 0; k0 < DD; k0 += 32) {
    if (k0 + 64 < DD) {
      prefetch_ro(wqT + (size_t)(tn * 32 + n) * DD + k0 + 64);
      prefetch_ro(wqT + (size_t)(tn * 32 + 16 + n) * DD + k0 + 64);
    }
    v16h fa0 = load_a_frag(xn, DD, tm * 32, k0);
    v16h fa1 = load_a_frag(xn, DD, tm * 32 + 16, k0);
    v16h fb0 = load_bt_frag(wqT, DD, k0, tn * 32);
    v16h fb1 = load_bt_frag(wqT, DD, k0, tn * 32 + 16);
    a00 = wmma_f16(fa0, fb0, a00);
    a01 = wmma_f16(fa0, fb1, a01);
    a10 = wmma_f16(fa1, fb0, a10);
    a11 = wmma_f16(fa1, fb1, a11);
  }
  qkv_store_tile(a00, tm * 32,      tn * 32,      q, k, vT);
  qkv_store_tile(a01, tm * 32,      tn * 32 + 16, q, k, vT);
  qkv_store_tile(a10, tm * 32 + 16, tn * 32,      q, k, vT);
  qkv_store_tile(a11, tm * 32 + 16, tn * 32 + 16, q, k, vT);
}

// ---------------- landmark means (groups of 32 tokens) ----------------
__global__ void k_landmarks(const _Float16* __restrict__ q, const _Float16* __restrict__ k,
                            _Float16* __restrict__ ql, _Float16* __restrict__ kl) {
  int dh = threadIdx.x;              // 64
  int m  = blockIdx.x;               // 256
  int bh = blockIdx.y;               // 32
  int sel = blockIdx.z;              // 0=q 1=k
  const _Float16* src = (sel ? k : q) + ((size_t)bh * NN + m * LL) * DH + dh;
  float s = 0.0f;
#pragma unroll 8
  for (int i = 0; i < LL; ++i) s += (float)src[(size_t)i * DH];
  _Float16* dst = sel ? kl : ql;
  dst[((size_t)bh * MM + m) * DH + dh] = (_Float16)(s * (1.0f / LL));
}

// ---------------- sim2 = q_l @ k_l^T, row softmax -> attn2 (f16) ----------------
__global__ void k_sim2_softmax(const _Float16* __restrict__ ql, const _Float16* __restrict__ kl,
                               _Float16* __restrict__ attn2) {
  __shared__ float sc[16 * MM];
  int bh = blockIdx.x, rt = blockIdx.y;
  const _Float16* qb = ql + (size_t)bh * MM * DH;
  const _Float16* kb = kl + (size_t)bh * MM * DH;
  int wave = threadIdx.x >> 5;
  v16h a0 = load_a_frag(qb, DH, rt * 16, 0);
  v16h a1 = load_a_frag(qb, DH, rt * 16, 32);
  for (int ct = wave * 2; ct < wave * 2 + 2; ++ct) {
    v8f acc = {};
    acc = wmma_f16(a0, load_bt_frag(kb, DH, 0,  ct * 16), acc);
    acc = wmma_f16(a1, load_bt_frag(kb, DH, 32, ct * 16), acc);
    store_d_f32(sc, MM, 0, ct * 16, acc);
  }
  __syncthreads();
  int t = threadIdx.x;
  if (t < 16) {
    float mx = -1e30f;
    for (int j = 0; j < MM; ++j) mx = fmaxf(mx, sc[t * MM + j]);
    float s = 0.0f;
    for (int j = 0; j < MM; ++j) { float e = __expf(sc[t * MM + j] - mx); sc[t * MM + j] = e; s += e; }
    float inv = 1.0f / s;
    _Float16* o = attn2 + ((size_t)bh * MM + rt * 16 + t) * MM;
    for (int j = 0; j < MM; ++j) o[j] = (_Float16)(sc[t * MM + j] * inv);
  }
}

// ---------------- pinv init: denom + z0 = x^T/denom (normal + transposed) ----------------
__global__ void k_pinv_init(const _Float16* __restrict__ attn2,
                            _Float16* __restrict__ z, _Float16* __restrict__ zT) {
  __shared__ float rs[MM], cs[MM], dsh;
  int bh = blockIdx.x, t = threadIdx.x;    // 256 threads
  const _Float16* xb = attn2 + (size_t)bh * MM * MM;
  float r = 0.0f, c = 0.0f;
  for (int j = 0; j < MM; ++j) {
    r += fabsf((float)xb[(size_t)t * MM + j]);
    c += fabsf((float)xb[(size_t)j * MM + t]);
  }
  rs[t] = r; cs[t] = c;
  __syncthreads();
  if (t == 0) {
    float mr = 0.0f, mc = 0.0f;
    for (int j = 0; j < MM; ++j) { mr = fmaxf(mr, rs[j]); mc = fmaxf(mc, cs[j]); }
    dsh = 1.0f / (mr * mc);
  }
  __syncthreads();
  float inv = dsh;
  _Float16* zb  = z  + (size_t)bh * MM * MM;
  _Float16* ztb = zT + (size_t)bh * MM * MM;
  for (int j = 0; j < MM; ++j) {
    zb[(size_t)t * MM + j]  = (_Float16)((float)xb[(size_t)j * MM + t] * inv);  // z0 = x^T/denom
    ztb[(size_t)t * MM + j] = (_Float16)((float)xb[(size_t)t * MM + j] * inv);  // z0^T = x/denom
  }
}

// ---------------- batched 256^3 GEMM (32x32 tile / wave), Newton-Schulz epilogues ----------------
__global__ void k_gemm256(const _Float16* __restrict__ A, const _Float16* __restrict__ Bt,
                          _Float16* __restrict__ out0, _Float16* __restrict__ out0t,
                          float scale, _Float16* __restrict__ out1t, float beta) {
  int bh = blockIdx.x;
  int wave = threadIdx.x >> 5;
  int tile = blockIdx.y * 8 + wave;        // 0..63
  int tm = tile >> 3, tn = tile & 7;       // 8x8 grid of 32x32 tiles
  const _Float16* Ab = A  + (size_t)bh * MM * MM;
  const _Float16* Bb = Bt + (size_t)bh * MM * MM;
  v8f a00 = {}, a01 = {}, a10 = {}, a11 = {};
#pragma unroll 2
  for (int k0 = 0; k0 < MM; k0 += 32) {
    v16h fa0 = load_a_frag(Ab, MM, tm * 32, k0);
    v16h fa1 = load_a_frag(Ab, MM, tm * 32 + 16, k0);
    v16h fb0 = load_bt_frag(Bb, MM, k0, tn * 32);
    v16h fb1 = load_bt_frag(Bb, MM, k0, tn * 32 + 16);
    a00 = wmma_f16(fa0, fb0, a00);
    a01 = wmma_f16(fa0, fb1, a01);
    a10 = wmma_f16(fa1, fb0, a10);
    a11 = wmma_f16(fa1, fb1, a11);
  }
  size_t base = (size_t)bh * MM * MM;
  v8f accs[4] = {a00, a01, a10, a11};
#pragma unroll
  for (int s = 0; s < 4; ++s) {
    int row0 = tm * 32 + (s >> 1) * 16;
    int col0 = tn * 32 + (s & 1) * 16;
    if (out0) {
      int lane = threadIdx.x & 31;
      int n = lane & 15, half = lane >> 4;
      F32x8 av; av.v = accs[s];
#pragma unroll
      for (int r = 0; r < 8; ++r)
        out0[base + (size_t)(row0 + half * 8 + r) * MM + col0 + n] = (_Float16)(av.f[r] * scale);
    }
    if (out0t) store_dT_f16(out0t + base, MM, row0, col0, accs[s], scale);
    if (out1t) store_dT_epi(out1t + base, MM, row0, col0, accs[s], beta);
  }
}

// ---------------- split-K flash: kvT = (softmax(q_l @ k^T) @ v)^T ----------------
// 8 waves each stream NN/8 keys with private online-softmax state; block combine.
__global__ void __launch_bounds__(256)
k_flash_kv(const _Float16* __restrict__ ql, const _Float16* __restrict__ kk,
           const _Float16* __restrict__ vT, _Float16* __restrict__ kvT) {
  __shared__ float sc[KS][16][32];
  __shared__ _Float16 p[KS][16][32];
  __shared__ float frow[KS][16];
  __shared__ float mw[KS][16], lw[KS][16];
  __shared__ float opart[KS][16][DH];
  int bh = blockIdx.x, rt = blockIdx.y;
  int wave = threadIdx.x >> 5;
  int lane = threadIdx.x & 31;
  int half = lane >> 4;
  const _Float16* qb = ql + (size_t)bh * MM * DH;
  const _Float16* kb = kk + (size_t)bh * NN * DH;
  const _Float16* vb = vT + (size_t)bh * DH * NN;   // [64][NN]
  v16h aq0 = load_a_frag(qb, DH, rt * 16, 0);
  v16h aq1 = load_a_frag(qb, DH, rt * 16, 32);
  v8f zero = {};
  v8f acc[4]; for (int t = 0; t < 4; ++t) acc[t] = zero;
  float mrun = -1e30f, lrun = 0.0f;
  int key_begin = wave * (NN / KS);
  for (int ch = 0; ch < (NN / KS) / 32; ++ch) {     // 32 chunks of 32 keys
    int key0 = key_begin + ch * 32;
    for (int ct = 0; ct < 2; ++ct) {
      v8f s = zero;
      s = wmma_f16(aq0, load_bt_frag(kb, DH, 0,  key0 + ct * 16), s);
      s = wmma_f16(aq1, load_bt_frag(kb, DH, 32, key0 + ct * 16), s);
      store_d_f32(&sc[wave][0][0], 32, 0, ct * 16, s);
    }
    __syncthreads();
    if (lane < 16) {
      float mc = -1e30f;
      for (int j = 0; j < 32; ++j) mc = fmaxf(mc, sc[wave][lane][j]);
      float mn = fmaxf(mrun, mc);
      float f = __expf(mrun - mn);
      float s = 0.0f;
      for (int j = 0; j < 32; ++j) {
        float e = __expf(sc[wave][lane][j] - mn);
        p[wave][lane][j] = (_Float16)e;
        s += e;
      }
      lrun = lrun * f + s;
      mrun = mn;
      frow[wave][lane] = f;
    }
    __syncthreads();
#pragma unroll
    for (int t = 0; t < 4; ++t) {
      F32x8 av; av.v = acc[t];
#pragma unroll
      for (int r = 0; r < 8; ++r) av.f[r] *= frow[wave][half * 8 + r];
      acc[t] = av.v;
    }
    v16h ap = load_a_frag(&p[wave][0][0], 32, 0, 0);
#pragma unroll
    for (int t = 0; t < 4; ++t)
      acc[t] = wmma_f16(ap, load_bt_frag(vb, NN, key0, t * 16), acc[t]);
  }
  // publish per-wave partial state
  if (lane < 16) { mw[wave][lane] = mrun; lw[wave][lane] = lrun; }
#pragma unroll
  for (int t = 0; t < 4; ++t) {
    F32x8 av; av.v = acc[t];
    int n = lane & 15;
#pragma unroll
    for (int r = 0; r < 8; ++r) opart[wave][half * 8 + r][t * 16 + n] = av.f[r];
  }
  __syncthreads();
  // combine across key splits: each thread handles 4 of the 16x64 outputs
  for (int idx = threadIdx.x; idx < 16 * DH; idx += 256) {
    int r = idx >> 6, c = idx & 63;
    float mg = -1e30f;
#pragma unroll
    for (int w = 0; w < KS; ++w) mg = fmaxf(mg, mw[w][r]);
    float lsum = 0.0f, osum = 0.0f;
#pragma unroll
    for (int w = 0; w < KS; ++w) {
      float f = __expf(mw[w][r] - mg);
      lsum += lw[w][r] * f;
      osum += opart[w][r][c] * f;
    }
    float res = (lsum > 0.0f) ? (osum / lsum) : 0.0f;
    kvT[((size_t)bh * DH + c) * MM + rt * 16 + r] = (_Float16)res;
  }
}

// ---------------- z2T = (attn2_inv @ kv)^T  ([256,256]@[256,64]) ----------------
__global__ void k_z2(const _Float16* __restrict__ z, const _Float16* __restrict__ kvT,
                     _Float16* __restrict__ z2T) {
  int bh = blockIdx.x;
  int wave = threadIdx.x >> 5;
  int tile = blockIdx.y * 8 + wave;        // 0..63
  int tm = tile >> 2, tn = tile & 3;
  const _Float16* Ab = z   + (size_t)bh * MM * MM;
  const _Float16* Bb = kvT + (size_t)bh * DH * MM;  // [64][256]
  v8f acc = {};
#pragma unroll 2
  for (int k0 = 0; k0 < MM; k0 += 32)
    acc = wmma_f16(load_a_frag(Ab, MM, tm * 16, k0), load_bt_frag(Bb, MM, k0, tn * 16), acc);
  store_dT_f16(z2T + (size_t)bh * DH * MM, MM, tm * 16, tn * 16, acc, 1.0f);
}

// ---------------- fused attn1 softmax-GEMM: softmax(q@k_l^T) @ z2 -> concat ----------------
__global__ void k_attn1(const _Float16* __restrict__ q, const _Float16* __restrict__ kl,
                        const _Float16* __restrict__ z2T, _Float16* __restrict__ concat) {
  __shared__ float sc[16 * MM];
  __shared__ _Float16 p[16 * MM];
  __shared__ float ored[16 * DH];
  int bh = blockIdx.x, rt = blockIdx.y;          // rt: 0..511
  const _Float16* qb = q  + (size_t)bh * NN * DH;
  const _Float16* kb = kl + (size_t)bh * MM * DH;
  int wave = threadIdx.x >> 5;
  v16h a0 = load_a_frag(qb, DH, rt * 16, 0);
  v16h a1 = load_a_frag(qb, DH, rt * 16, 32);
  for (int ct = wave * 2; ct < wave * 2 + 2; ++ct) {
    v8f acc = {};
    acc = wmma_f16(a0, load_bt_frag(kb, DH, 0,  ct * 16), acc);
    acc = wmma_f16(a1, load_bt_frag(kb, DH, 32, ct * 16), acc);
    store_d_f32(sc, MM, 0, ct * 16, acc);
  }
  __syncthreads();
  int t = threadIdx.x;
  if (t < 16) {
    float mx = -1e30f;
    for (int j = 0; j < MM; ++j) mx = fmaxf(mx, sc[t * MM + j]);
    float s = 0.0f;
    for (int j = 0; j < MM; ++j) { float e = __expf(sc[t * MM + j] - mx); s += e; sc[t * MM + j] = e; }
    float inv = 1.0f / s;
    for (int j = 0; j < MM; ++j) p[t * MM + j] = (_Float16)(sc[t * MM + j] * inv);
  }
  __syncthreads();
  // phase 2: all 8 waves; split K in halves: wave = ks*4 + ct
  {
    const _Float16* z2b = z2T + (size_t)bh * DH * MM;  // [64][256]
    int ct = wave & 3, ks = wave >> 2;
    v8f acc = {};
#pragma unroll
    for (int k0 = ks * 128; k0 < ks * 128 + 128; k0 += 32)
      acc = wmma_f16(load_a_frag(p, MM, 0, k0), load_bt_frag(z2b, MM, k0, ct * 16), acc);
    if (ks == 1) store_d_f32(ored, DH, 0, ct * 16, acc);
    __syncthreads();
    if (ks == 0) {
      int lane = threadIdx.x & 31;
      int n = lane & 15, half = lane >> 4;
      int bi = bh >> 3, hh = bh & 7;
      F32x8 av; av.v = acc;
#pragma unroll
      for (int r = 0; r < 8; ++r) {
        int tok = rt * 16 + half * 8 + r;
        float val = av.f[r] + ored[(half * 8 + r) * DH + ct * 16 + n];
        concat[((size_t)bi * NN + tok) * INNER + hh * DH + ct * 16 + n] = (_Float16)val;
      }
    }
  }
}

// ---------------- depthwise conv residual (kernel 33 along seq) added into concat ----------------
__global__ void k_conv_add(const _Float16* __restrict__ vT, const float* __restrict__ rw,
                           _Float16* __restrict__ concat) {
  size_t i = (size_t)blockIdx.x * 256 + threadIdx.x;  // over BH*DH*NN, tok innermost
  int tok = (int)(i & (NN - 1));
  int dh  = (int)((i >> 13) & 63);
  int bh  = (int)(i >> 19);
  int bi = bh >> 3, hh = bh & 7;
  const _Float16* vrow = vT + ((size_t)bh * DH + dh) * NN;  // contiguous in tok
  float s = 0.0f;
#pragma unroll
  for (int kk = 0; kk < 33; ++kk) {
    int tt = tok + kk - 16;
    if (tt >= 0 && tt < NN) s += (float)vrow[tt] * rw[hh * 33 + kk];
  }
  size_t o = ((size_t)bi * NN + tok) * INNER + hh * DH + dh;
  concat[o] = (_Float16)((float)concat[o] + s);
}

// ---------------- final (32x32 tile / wave): concat @ w_out + b_out + x -> out ----------------
__global__ void k_final(const _Float16* __restrict__ concat, const _Float16* __restrict__ woT,
                        const float* __restrict__ bout, const float* __restrict__ x,
                        float* __restrict__ out) {
  int wave = threadIdx.x >> 5;
  int tile = blockIdx.x * 8 + wave;
  const int TN = DD / 32;                  // 16
  int tm = tile / TN, tn = tile % TN;
  int lane = threadIdx.x & 31;
  int n = lane & 15, half = lane >> 4;
  v8f a00 = {}, a01 = {}, a10 = {}, a11 = {};
#pragma unroll 4
  for (int k0 = 0; k0 < INNER; k0 += 32) {
    if (k0 + 64 < INNER) {
      prefetch_ro(woT + (size_t)(tn * 32 + n) * DD + k0 + 64);
      prefetch_ro(woT + (size_t)(tn * 32 + 16 + n) * DD + k0 + 64);
    }
    v16h fa0 = load_a_frag(concat, INNER, tm * 32, k0);
    v16h fa1 = load_a_frag(concat, INNER, tm * 32 + 16, k0);
    v16h fb0 = load_bt_frag(woT, DD, k0, tn * 32);
    v16h fb1 = load_bt_frag(woT, DD, k0, tn * 32 + 16);
    a00 = wmma_f16(fa0, fb0, a00);
    a01 = wmma_f16(fa0, fb1, a01);
    a10 = wmma_f16(fa1, fb0, a10);
    a11 = wmma_f16(fa1, fb1, a11);
  }
  v8f accs[4] = {a00, a01, a10, a11};
#pragma unroll
  for (int s = 0; s < 4; ++s) {
    int row0 = tm * 32 + (s >> 1) * 16;
    int col = tn * 32 + (s & 1) * 16 + n;
    F32x8 av; av.v = accs[s];
#pragma unroll
    for (int r = 0; r < 8; ++r) {
      size_t row = (size_t)row0 + half * 8 + r;
      out[row * DD + col] = av.f[r] + bout[col] + x[row * DD + col];
    }
  }
}

extern "C" void kernel_launch(void* const* d_in, const int* in_sizes, int n_in,
                              void* d_out, int out_size, void* d_ws, size_t ws_size,
                              hipStream_t stream) {
  const float* x     = (const float*)d_in[0];
  const float* ln_g  = (const float*)d_in[1];
  const float* ln_b  = (const float*)d_in[2];
  const float* w_qkv = (const float*)d_in[3];
  const float* w_out = (const float*)d_in[4];
  const float* b_out = (const float*)d_in[5];
  const float* res_w = (const float*)d_in[6];
  float* out = (float*)d_out;

  char* p = (char*)d_ws;
  auto alloc = [&](size_t bytes) -> char* {
    char* r = p; p += (bytes + 255) & ~(size_t)255; return r;
  };
  _Float16* xn   = (_Float16*)alloc((size_t)BN * DD * 2);      // reused as concat later
  _Float16* wqT  = (_Float16*)alloc((size_t)DD * NQKV * 2);
  _Float16* woT  = (_Float16*)alloc((size_t)DD * DD * 2);
  _Float16* q16  = (_Float16*)alloc((size_t)BH * NN * DH * 2);
  _Float16* k16  = (_Float16*)alloc((size_t)BH * NN * DH * 2);
  _Float16* vT   = (_Float16*)alloc((size_t)BH * NN * DH * 2);
  _Float16* ql16 = (_Float16*)alloc((size_t)BH * MM * DH * 2);
  _Float16* kl16 = (_Float16*)alloc((size_t)BH * MM * DH * 2);
  _Float16* at2  = (_Float16*)alloc((size_t)BH * MM * MM * 2);
  _Float16* za   = (_Float16*)alloc((size_t)BH * MM * MM * 2);
  _Float16* zaT  = (_Float16*)alloc((size_t)BH * MM * MM * 2);
  _Float16* zb   = (_Float16*)alloc((size_t)BH * MM * MM * 2);
  _Float16* zbT  = (_Float16*)alloc((size_t)BH * MM * MM * 2);
  _Float16* xz   = (_Float16*)alloc((size_t)BH * MM * MM * 2);
  _Float16* taT  = (_Float16*)alloc((size_t)BH * MM * MM * 2);
  _Float16* tbT  = (_Float16*)alloc((size_t)BH * MM * MM * 2);
  _Float16* kvT  = (_Float16*)alloc((size_t)BH * MM * DH * 2);
  _Float16* z2T  = (_Float16*)alloc((size_t)BH * MM * DH * 2);
  _Float16* concat = xn;  // xn no longer needed after QKV GEMM

  k_layernorm<<<BN, 256, 0, stream>>>(x, ln_g, ln_b, xn);
  k_cvt_w<<<(DD * NQKV + 255) / 256, 256, 0, stream>>>(w_qkv, w_out, wqT, woT);
  k_qkv_gemm<<<(BN / 32) * (NQKV / 32) / 8, 256, 0, stream>>>(xn, wqT, q16, k16, vT);
  k_landmarks<<<dim3(MM, BH, 2), 64, 0, stream>>>(q16, k16, ql16, kl16);
  k_sim2_softmax<<<dim3(BH, MM / 16), 256, 0, stream>>>(ql16, kl16, at2);
  k_pinv_init<<<BH, 256, 0, stream>>>(at2, za, zaT);

  _Float16* zc = za;  _Float16* zcT = zaT;
  _Float16* zn = zb;  _Float16* znT = zbT;
  for (int it = 0; it < 6; ++it) {
    // xz = x@z ; taT = (7I - xz)^T
    k_gemm256<<<dim3(BH, 8), 256, 0, stream>>>(at2, zcT, xz, (_Float16*)nullptr, 1.0f, taT, 7.0f);
    // tbT = (15I - xz@t)^T
    k_gemm256<<<dim3(BH, 8), 256, 0, stream>>>(xz, taT, (_Float16*)nullptr, (_Float16*)nullptr, 1.0f, tbT, 15.0f);
    // taT = (13I - xz@t)^T
    k_gemm256<<<dim3(BH, 8), 256, 0, stream>>>(xz, tbT, (_Float16*)nullptr, (_Float16*)nullptr, 1.0f, taT, 13.0f);
    // z' = 0.25 * z@t  (both layouts)
    k_gemm256<<<dim3(BH, 8), 256, 0, stream>>>(zc, taT, zn, znT, 0.25f, (_Float16*)nullptr, 0.0f);
    _Float16* t0 = zc;  zc  = zn;  zn  = t0;
    _Float16* t1 = zcT; zcT = znT; znT = t1;
  }

  k_flash_kv<<<dim3(BH, MM / 16), 256, 0, stream>>>(ql16, k16, vT, kvT);
  k_z2<<<dim3(BH, 8), 256, 0, stream>>>(zc, kvT, z2T);
  k_attn1<<<dim3(BH, NN / 16), 256, 0, stream>>>(q16, kl16, z2T, concat);
  k_conv_add<<<(BH * DH * NN) / 256, 256, 0, stream>>>(vT, res_w, concat);
  k_final<<<(BN / 32) * (DD / 32) / 8, 256, 0, stream>>>(concat, woT, b_out, x, out);
}
